// Graphsage_Encoder_21921513079204
// MI455X (gfx1250) — matrix operand order repaired and checked by reading, not compiled, not run on hardware
//
#include <hip/hip_runtime.h>

// ---------------------------------------------------------------------------
// GraphSAGE encoder for MI455X (gfx1250), wave32 + WMMA bf16.
//
// Key transforms vs reference:
//  * proj is linear->linear with no activation in between on every branch, so
//    we fold W_c = W_hop0 @ W_proj ([256,256]) once per branch (tiny kernel)
//    and skip the 262144x256x128 proj GEMM + its 134MB intermediate entirely.
//  * final layer: hop1/2/3 are per-batch broadcasts, so
//    out = hop0 @ Wfin[:, :256]^T + cbias[b],  cbias = bfin + hopcat @ Wfin[:,256:]^T
//    (8x2048 constant), collapsing the 2048-K GEMM to K=256.
//  * BatchNorm (training mode) = deterministic 2-stage column sum/sumsq
//    reduction; BN(+ReLU) is fused into the segment-mean / row-mean readers.
//  * All GEMMs run through one bf16 WMMA kernel (v_wmma_f32_16x16x32_bf16),
//    converting f32 -> bf16 (RNE) while staging tiles into LDS.
// ---------------------------------------------------------------------------

typedef __attribute__((ext_vector_type(16))) __bf16 v16bf;
typedef __attribute__((ext_vector_type(8)))  float  v8f;

__device__ __forceinline__ unsigned short f2bf(float x) {
  union { float f; unsigned u; } c; c.f = x;
  unsigned r = c.u + 0x7FFFu + ((c.u >> 16) & 1u);   // round-to-nearest-even
  return (unsigned short)(r >> 16);
}

union FragBF { unsigned u[8]; v16bf v; };

// C[M,N] = A[M,K] @ W[N,K]^T  (+bias).  bias_mode: 0 none, 1 bias[n],
// 2 bias[(m/rpb)*N + n].  M,N multiples of 128; K multiple of 32.
__global__ __launch_bounds__(256)
void wmma_gemm_bf16(const float* __restrict__ A, const float* __restrict__ W,
                    const float* __restrict__ bias, float* __restrict__ C,
                    int M, int N, int K, int lda, int ldw, int ldc,
                    int bias_mode, int rpb) {
  __shared__ __align__(16) unsigned short As[128][40];   // 128 rows x 32 K (bf16), padded
  __shared__ __align__(16) unsigned short Ws[128][40];   // 128 cols x 32 K (bf16)

  const int t    = threadIdx.x;
  const int m0   = blockIdx.y * 128;
  const int n0   = blockIdx.x * 128;
  const int lane = t & 31;
  const int wv   = t >> 5;
  const int mbase = (wv & 3) * 32;   // wave's 32 output rows
  const int nbase = (wv >> 2) * 64;  // wave's 64 output cols
  const int rlo  = lane & 15;
  const int hi   = lane >> 4;

  v8f acc[2][4];
  #pragma unroll
  for (int i = 0; i < 2; ++i)
    #pragma unroll
    for (int j = 0; j < 4; ++j) acc[i][j] = {};

  for (int k0 = 0; k0 < K; k0 += 32) {
    // Stage A tile (128x32) and W tile (128x32) as bf16 pairs (one b32 LDS
    // store per pair -> clean ds_store_b32 traffic).
    #pragma unroll
    for (int i = 0; i < 8; ++i) {
      int e = i * 256 + t;
      int r = e >> 4, pc = (e & 15) * 2;
      const float* src = A + (size_t)(m0 + r) * lda + k0 + pc;
      unsigned p = (unsigned)f2bf(src[0]) | ((unsigned)f2bf(src[1]) << 16);
      *(unsigned*)&As[r][pc] = p;
    }
    #pragma unroll
    for (int i = 0; i < 8; ++i) {
      int e = i * 256 + t;
      int r = e >> 4, pc = (e & 15) * 2;
      const float* src = W + (size_t)(n0 + r) * ldw + k0 + pc;
      unsigned p = (unsigned)f2bf(src[0]) | ((unsigned)f2bf(src[1]) << 16);
      *(unsigned*)&Ws[r][pc] = p;
    }
    if (k0 + 32 < K) {  // speculative prefetch of next K tile (global_prefetch_b8)
      __builtin_prefetch(A + (size_t)(m0 + (t & 127)) * lda + k0 + 32, 0, 1);
      __builtin_prefetch(W + (size_t)(n0 + (t & 127)) * ldw + k0 + 32, 0, 1);
    }
    __syncthreads();

    // A fragments: 16x32 bf16, per ISA layout (lanes 0-15: K 0..7 & 16..23,
    // lanes 16-31: +8), one row per lane.
    FragBF afr[2];
    #pragma unroll
    for (int i = 0; i < 2; ++i) {
      int row = mbase + i * 16 + rlo;
      #pragma unroll
      for (int v = 0; v < 8; ++v) {
        int kk = ((v < 4) ? (2 * v) : (16 + 2 * (v - 4))) + hi * 8;
        afr[i].u[v] = *(const unsigned*)&As[row][kk];
      }
    }
    // B fragments: 32x16 bf16 (B[k][n] = W[n][k]); lane = column, lanes 0-15
    // hold K 0..7 & 16..23, lanes 16-31: +8.
    FragBF bfr[4];
    #pragma unroll
    for (int j = 0; j < 4; ++j) {
      int col = nbase + j * 16 + rlo;
      #pragma unroll
      for (int v = 0; v < 8; ++v)
        bfr[j].u[v] = *(const unsigned*)&Ws[col][((v < 4) ? (2 * v) : (16 + 2 * (v - 4))) + hi * 8];
    }
    #pragma unroll
    for (int i = 0; i < 2; ++i)
      #pragma unroll
      for (int j = 0; j < 4; ++j)
        acc[i][j] = __builtin_amdgcn_wmma_f32_16x16x32_bf16(
            false, afr[i].v, false, bfr[j].v, (short)0, acc[i][j], false, false);
    __syncthreads();
  }

  // Epilogue: D layout -> lane holds N=(lane&15); VGPR v holds M = v + 8*hi.
  #pragma unroll
  for (int i = 0; i < 2; ++i)
    #pragma unroll
    for (int j = 0; j < 4; ++j)
      #pragma unroll
      for (int v = 0; v < 8; ++v) {
        int m = m0 + mbase + i * 16 + hi * 8 + v;
        int n = n0 + nbase + j * 16 + rlo;
        float val = acc[i][j][v];
        if (bias_mode == 1)      val += bias[n];
        else if (bias_mode == 2) val += bias[(size_t)(m / rpb) * N + n];
        if (m < M && n < N) C[(size_t)m * ldc + n] = val;
      }
}

// Fold W_c = Whop[256,128] @ Wproj[128,256]; bc = Whop@bproj + bhop.
__global__ void combine_w(const float* __restrict__ Whop, const float* __restrict__ bhop,
                          const float* __restrict__ Wproj, const float* __restrict__ bproj,
                          float* __restrict__ Wc, float* __restrict__ bc) {
  int i = blockIdx.y;                       // 0..255 (fan_out)
  int j = threadIdx.x;                      // 0..255 (fan_in = FEAT)
  float s = 0.f;
  for (int k = 0; k < 128; ++k) s += Whop[i * 128 + k] * Wproj[k * 256 + j];
  Wc[i * 256 + j] = s;
  if (j == 0) {
    float sb = bhop[i];
    for (int k = 0; k < 128; ++k) sb += Whop[i * 128 + k] * bproj[k];
    bc[i] = sb;
  }
}

// Column partial sums / sumsq over a row stripe (deterministic 2-stage BN stats).
__global__ void colstats_partial(const float* __restrict__ Z, int M, int F, int rowsPerBlk,
                                 float* __restrict__ psum, float* __restrict__ psumsq) {
  int f = blockIdx.x * blockDim.x + threadIdx.x;
  if (f >= F) return;
  int rb = blockIdx.y;
  int r0 = rb * rowsPerBlk;
  int r1 = r0 + rowsPerBlk; if (r1 > M) r1 = M;
  float s = 0.f, s2 = 0.f;
  for (int r = r0; r < r1; ++r) {
    float v = Z[(size_t)r * F + f];
    s += v; s2 += v * v;
  }
  psum[(size_t)rb * F + f] = s;
  psumsq[(size_t)rb * F + f] = s2;
}

__global__ void colstats_finalize(const float* __restrict__ psum, const float* __restrict__ psumsq,
                                  int nblk, int F, float invM,
                                  float* __restrict__ mean, float* __restrict__ invstd) {
  int f = blockIdx.x * blockDim.x + threadIdx.x;
  if (f >= F) return;
  float s = 0.f, s2 = 0.f;
  for (int b = 0; b < nblk; ++b) { s += psum[(size_t)b * F + f]; s2 += psumsq[(size_t)b * F + f]; }
  float m = s * invM;
  float v = s2 * invM - m * m; if (v < 0.f) v = 0.f;
  mean[f] = m;
  invstd[f] = rsqrtf(v + 1e-5f);
}

// Per-batch exclusive prefix of counts -> offsets [B][P+1].
__global__ void seg_offsets(const int* __restrict__ cnt, int P, int* __restrict__ off) {
  int b = blockIdx.x;
  if (threadIdx.x != 0) return;
  int acc = 0;
  for (int p = 0; p < P; ++p) { off[b * (P + 1) + p] = acc; acc += cnt[b * P + p]; }
  off[b * (P + 1) + P] = acc;
}

// Segment mean of act(BN(Z)) : Z [B,S,F] -> out [B,P,F]; one block per (b,p).
__global__ void agg_bn(const float* __restrict__ Z, const int* __restrict__ off,
                       int S, int P, int F,
                       const float* __restrict__ mean, const float* __restrict__ invstd,
                       const float* __restrict__ g, const float* __restrict__ be,
                       int relu, float* __restrict__ out) {
  int bp = blockIdx.x;
  int b = bp / P, p = bp - b * P;
  int s0 = off[b * (P + 1) + p];
  int s1 = off[b * (P + 1) + p + 1];
  int cnt = s1 - s0;
  float inv = 1.0f / (float)(cnt > 1 ? cnt : 1);
  for (int f = threadIdx.x; f < F; f += blockDim.x) {
    float sc = invstd[f] * g[f];
    float sh = be[f] - mean[f] * sc;
    float s = 0.f;
    for (int r = s0; r < s1; ++r) {
      float v = Z[((size_t)b * S + r) * F + f] * sc + sh;
      if (relu) v = fmaxf(v, 0.f);
      s += v;
    }
    out[((size_t)b * P + p) * F + f] = s * inv;
  }
}

// Row mean of act(BN(Z)) over S -> dst[b*dstLd + dstOff + f]  (hop vectors).
__global__ void rowmean_bn(const float* __restrict__ Z, int S, int F,
                           const float* __restrict__ mean, const float* __restrict__ invstd,
                           const float* __restrict__ g, const float* __restrict__ be,
                           int relu, float* __restrict__ dst, int dstLd, int dstOff) {
  int b = blockIdx.y;
  int f = blockIdx.x * blockDim.x + threadIdx.x;
  if (f >= F) return;
  float sc = invstd[f] * g[f];
  float sh = be[f] - mean[f] * sc;
  float s = 0.f;
  for (int r = 0; r < S; ++r) {
    float v = Z[((size_t)b * S + r) * F + f] * sc + sh;
    if (relu) v = fmaxf(v, 0.f);
    s += v;
  }
  dst[(size_t)b * dstLd + dstOff + f] = s / (float)S;
}

// In-place BN (hop0).
__global__ void bn_apply(float* __restrict__ Z, size_t total, int F,
                         const float* __restrict__ mean, const float* __restrict__ invstd,
                         const float* __restrict__ g, const float* __restrict__ be) {
  size_t i = (size_t)blockIdx.x * blockDim.x + threadIdx.x;
  if (i >= total) return;
  int f = (int)(i % F);
  float sc = invstd[f] * g[f];
  float sh = be[f] - mean[f] * sc;
  Z[i] = Z[i] * sc + sh;
}

// cbias[b,n] = b_fin[n] + sum_k hopcat[b,k] * Wfin[n, 256+k]   (k < 1792)
__global__ void cbias_kernel(const float* __restrict__ hopcat, const float* __restrict__ Wfin,
                             const float* __restrict__ bfin, float* __restrict__ cb) {
  int b = blockIdx.y;
  int n = blockIdx.x * blockDim.x + threadIdx.x;
  if (n >= 2048) return;
  float s = bfin[n];
  const float* w = Wfin + (size_t)n * 2048 + 256;
  const float* h = hopcat + b * 1792;
  for (int k = 0; k < 1792; ++k) s += h[k] * w[k];
  cb[b * 2048 + n] = s;
}

extern "C" void kernel_launch(void* const* d_in, const int* in_sizes, int n_in,
                              void* d_out, int out_size, void* d_ws, size_t ws_size,
                              hipStream_t stream) {
  (void)in_sizes; (void)n_in; (void)out_size; (void)ws_size;
  const int B = 8, N0 = 1024, S1 = 2048, P2 = 512, P3 = 4096, S3 = 32768;

  const float* nodes3 = (const float*)d_in[0];
  const float* nodes2 = (const float*)d_in[1];
  const float* nodes1 = (const float*)d_in[2];
  const float* nodes0 = (const float*)d_in[3];
  const int*   cnt3   = (const int*)d_in[4];
  const int*   cnt2   = (const int*)d_in[5];
  // params in dict insertion order
  const float* W_proj = (const float*)d_in[6];  const float* b_proj = (const float*)d_in[7];
  const float* W_3_0  = (const float*)d_in[8];  const float* b_3_0  = (const float*)d_in[9];
  const float* W_3_1  = (const float*)d_in[10]; const float* b_3_1  = (const float*)d_in[11];
  const float* W_3_2  = (const float*)d_in[12]; const float* b_3_2  = (const float*)d_in[13];
  const float* W_2_0  = (const float*)d_in[14]; const float* b_2_0  = (const float*)d_in[15];
  const float* W_2_1  = (const float*)d_in[16]; const float* b_2_1  = (const float*)d_in[17];
  const float* W_1_0  = (const float*)d_in[18]; const float* b_1_0  = (const float*)d_in[19];
  const float* W_0_0  = (const float*)d_in[20]; const float* b_0_0  = (const float*)d_in[21];
  const float* W_fin  = (const float*)d_in[22]; const float* b_fin  = (const float*)d_in[23];
  const float* g_3_0  = (const float*)d_in[24]; const float* be_3_0 = (const float*)d_in[25];
  const float* g_3_1  = (const float*)d_in[26]; const float* be_3_1 = (const float*)d_in[27];
  const float* g_3_2  = (const float*)d_in[28]; const float* be_3_2 = (const float*)d_in[29];
  const float* g_2_0  = (const float*)d_in[30]; const float* be_2_0 = (const float*)d_in[31];
  const float* g_2_1  = (const float*)d_in[32]; const float* be_2_1 = (const float*)d_in[33];
  const float* g_1_0  = (const float*)d_in[34]; const float* be_1_0 = (const float*)d_in[35];
  const float* g_0_0  = (const float*)d_in[36]; const float* be_0_0 = (const float*)d_in[37];

  // ---- workspace layout ----
  char* w = (char*)d_ws;
  float* Z    = (float*)(w);                          // 268,435,456 B (biggest linear out)
  float* Abuf = (float*)(w + 268435456ULL);           //  33,554,432 B (agg outputs)
  float* H0   = (float*)(w + 301989888ULL);           //   8,388,608 B (hop0, kept for fin)
  float* WC3  = (float*)(w + 310378496ULL);
  float* WC2  = WC3 + 65536; float* WC1 = WC2 + 65536; float* WC0 = WC1 + 65536;
  float* BC3  = WC0 + 65536; float* BC2 = BC3 + 256; float* BC1 = BC2 + 256; float* BC0 = BC1 + 256;
  float* HOPCAT = BC0 + 256;          // [8,1792]: hop1 @0, hop2 @256, hop3 @768
  float* CB   = HOPCAT + 8 * 1792;    // [8,2048]
  float* PS   = CB + 8 * 2048;        // 32768 partial sums
  float* PS2  = PS + 32768;           // 32768 partial sumsq
  float* MEAN = PS2 + 32768;          // 1024
  float* ISTD = MEAN + 1024;          // 1024
  int*   OFF3 = (int*)(ISTD + 1024);  // 8*(4096+1)
  int*   OFF2 = OFF3 + 8 * (P3 + 1);  // 8*(512+1)

  auto gemm = [&](const float* A, const float* Wt, const float* bias, float* C,
                  int M, int N, int K, int lda, int ldw, int ldc, int bmode, int rpb) {
    dim3 grid(N / 128, M / 128);
    wmma_gemm_bf16<<<grid, 256, 0, stream>>>(A, Wt, bias, C, M, N, K, lda, ldw, ldc, bmode, rpb);
  };
  auto stats = [&](const float* Zp, int M, int F) {
    const int rows = 2048;
    int nblk = (M + rows - 1) / rows;   // nblk*F <= 32768 for all layers
    colstats_partial<<<dim3((F + 255) / 256, nblk), 256, 0, stream>>>(Zp, M, F, rows, PS, PS2);
    colstats_finalize<<<dim3((F + 255) / 256), 256, 0, stream>>>(PS, PS2, nblk, F, 1.0f / (float)M, MEAN, ISTD);
  };

  // ---- prep: segment offsets + folded proj weights ----
  seg_offsets<<<B, 32, 0, stream>>>(cnt3, P3, OFF3);
  seg_offsets<<<B, 32, 0, stream>>>(cnt2, P2, OFF2);
  combine_w<<<dim3(1, 256), 256, 0, stream>>>(W_3_0, b_3_0, W_proj, b_proj, WC3, BC3);
  combine_w<<<dim3(1, 256), 256, 0, stream>>>(W_2_0, b_2_0, W_proj, b_proj, WC2, BC2);
  combine_w<<<dim3(1, 256), 256, 0, stream>>>(W_1_0, b_1_0, W_proj, b_proj, WC1, BC1);
  combine_w<<<dim3(1, 256), 256, 0, stream>>>(W_0_0, b_0_0, W_proj, b_proj, WC0, BC0);

  // ---- hop-3 branch: S3 -> P3 -> P2 -> mean ----
  gemm(nodes3, WC3, BC3, Z, B * S3, 256, 256, 256, 256, 256, 1, 0);         // [262144,256]
  stats(Z, B * S3, 256);
  agg_bn<<<B * P3, 256, 0, stream>>>(Z, OFF3, S3, P3, 256, MEAN, ISTD, g_3_0, be_3_0, 1, Abuf);
  gemm(Abuf, W_3_1, b_3_1, Z, B * P3, 512, 256, 256, 256, 512, 1, 0);       // [32768,512]
  stats(Z, B * P3, 512);
  agg_bn<<<B * P2, 256, 0, stream>>>(Z, OFF2, P3, P2, 512, MEAN, ISTD, g_3_1, be_3_1, 1, Abuf);
  gemm(Abuf, W_3_2, b_3_2, Z, B * P2, 1024, 512, 512, 512, 1024, 1, 0);     // [4096,1024]
  stats(Z, B * P2, 1024);
  rowmean_bn<<<dim3(4, B), 256, 0, stream>>>(Z, P2, 1024, MEAN, ISTD, g_3_2, be_3_2, 0, HOPCAT, 1792, 768);

  // ---- hop-2 branch: P3 -> P2 -> mean ----
  gemm(nodes2, WC2, BC2, Z, B * P3, 256, 256, 256, 256, 256, 1, 0);         // [32768,256]
  stats(Z, B * P3, 256);
  agg_bn<<<B * P2, 256, 0, stream>>>(Z, OFF2, P3, P2, 256, MEAN, ISTD, g_2_0, be_2_0, 1, Abuf);
  gemm(Abuf, W_2_1, b_2_1, Z, B * P2, 512, 256, 256, 256, 512, 1, 0);       // [4096,512]
  stats(Z, B * P2, 512);
  rowmean_bn<<<dim3(2, B), 256, 0, stream>>>(Z, P2, 512, MEAN, ISTD, g_2_1, be_2_1, 0, HOPCAT, 1792, 256);

  // ---- hop-1 branch: mean ----
  gemm(nodes1, WC1, BC1, Z, B * S1, 256, 256, 256, 256, 256, 1, 0);         // [16384,256]
  stats(Z, B * S1, 256);
  rowmean_bn<<<dim3(1, B), 256, 0, stream>>>(Z, S1, 256, MEAN, ISTD, g_1_0, be_1_0, 0, HOPCAT, 1792, 0);

  // ---- hop-0 (self) ----
  gemm(nodes0, WC0, BC0, H0, B * N0, 256, 256, 256, 256, 256, 1, 0);        // [8192,256]
  stats(H0, B * N0, 256);
  bn_apply<<<(B * N0 * 256) / 256, 256, 0, stream>>>(H0, (size_t)B * N0 * 256, 256, MEAN, ISTD, g_0_0, be_0_0);

  // ---- final: out = hop0 @ Wfin[:, :256]^T + cbias[b] ----
  cbias_kernel<<<dim3(8, B), 256, 0, stream>>>(HOPCAT, W_fin, b_fin, CB);
  gemm(H0, W_fin, CB, (float*)d_out, B * N0, 2048, 256, 256, 2048, 2048, 2, N0);
}
